// GraphMAEModel_58995670778276
// MI455X (gfx1250) — compile-verified, matrix operand order checked by reading, not run on hardware
//
#include <hip/hip_runtime.h>
#include <hip/hip_bf16.h>

// ---------- types ----------
typedef __attribute__((ext_vector_type(8)))  __bf16 v8bf;
typedef __attribute__((ext_vector_type(16))) __bf16 v16bf;
typedef __attribute__((ext_vector_type(8)))  float  v8f;

// ---------- helpers ----------
__device__ __forceinline__ unsigned short f2bf(float f) {
    unsigned int u = __float_as_uint(f);
    unsigned int r = u + 0x7FFFu + ((u >> 16) & 1u);   // round-to-nearest-even
    return (unsigned short)(r >> 16);
}
__device__ __forceinline__ float bf2f(unsigned short u) {
    return __uint_as_float(((unsigned int)u) << 16);
}
// Load a 16-bf16 WMMA fragment: two 16B chunks, p[0..7] -> elements 0..7 and p[16..23] -> elements 8..15.
__device__ __forceinline__ v16bf load_frag(const unsigned short* p) {
    v8bf lo = *reinterpret_cast<const v8bf*>(p);
    v8bf hi = *reinterpret_cast<const v8bf*>(p + 16);
    return __builtin_shufflevector(lo, hi, 0,1,2,3,4,5,6,7,8,9,10,11,12,13,14,15);
}

// ---------- weight f32 -> swizzled bf16 [k/32][n][k%32] ----------
__global__ __launch_bounds__(256) void conv_w_kernel(const float* __restrict__ W,
                                                     unsigned short* __restrict__ out) {
    int tid = blockIdx.x * 256 + threadIdx.x;          // 65536 elements, W row-major [K=256][N=256]
    int k = tid >> 8, n = tid & 255;
    out[(k >> 5) * 8192 + n * 32 + (k & 31)] = f2bf(W[tid]);
}

// ---------- encoder first layer (K=2, trivial) -> bf16 ----------
__global__ __launch_bounds__(256) void enc_pre_kernel(const float* __restrict__ x,
                                                      const float* __restrict__ We1,
                                                      const float* __restrict__ be1,
                                                      unsigned short* __restrict__ pre) {
    int n = blockIdx.x, j = threadIdx.x;
    float v = x[n * 2] * We1[j] + x[n * 2 + 1] * We1[256 + j] + be1[j];
    pre[(size_t)n * 256 + j] = f2bf(fmaxf(v, 0.0f));
}

// ---------- WMMA GEMM: C[M,256] = act(A[M,256] @ B[256,256] + bias) ----------
// A: bf16 row-major; Bsw: bf16 swizzled [kc][n][kk]; outputs f32 and/or bf16 (nullable).
__global__ __launch_bounds__(256) void gemm256_wmma_kernel(
        const unsigned short* __restrict__ A,
        const unsigned short* __restrict__ Bsw,
        const float* __restrict__ bias,            // nullable
        float* __restrict__ Cf,                    // nullable
        unsigned short* __restrict__ Cbf,          // nullable
        int M, int relu_flag) {
    const int lane = threadIdx.x & 31;
    const int wave = threadIdx.x >> 5;
    const int m0 = blockIdx.x * 32 + (wave >> 2) * 16;   // 16-row tile per wave
    if (m0 >= M) return;
    const int n0 = (wave & 3) * 64;                      // 64-col group per wave
    const int lr = lane & 15;
    const int lh = lane >> 4;

    v8f acc[4] = {};
    const unsigned short* Arow = A + (size_t)(m0 + lr) * 256 + lh * 8;
    const unsigned short* Bcol = Bsw + (size_t)(n0 + lr) * 32 + lh * 8;

    #pragma unroll
    for (int kc = 0; kc < 8; ++kc) {
        v16bf af = load_frag(Arow + kc * 32);
        const unsigned short* Bb = Bcol + kc * 8192;
        #pragma unroll
        for (int t = 0; t < 4; ++t) {
            v16bf bfv = load_frag(Bb + t * 512);         // +16 cols => +512 elements
            acc[t] = __builtin_amdgcn_wmma_f32_16x16x32_bf16(
                         false, af, false, bfv, (short)0, acc[t], false, false);
        }
    }

    #pragma unroll
    for (int t = 0; t < 4; ++t) {
        const int col = n0 + t * 16 + lr;
        const float bv = bias ? bias[col] : 0.0f;
        #pragma unroll
        for (int r = 0; r < 8; ++r) {
            const int row = m0 + lh * 8 + r;             // C layout: VGPR r -> row r + 8*(lane/16)
            float v = acc[t][r] + bv;
            if (relu_flag) v = fmaxf(v, 0.0f);
            if (Cf)  Cf[(size_t)row * 256 + col] = v;
            if (Cbf) Cbf[(size_t)row * 256 + col] = f2bf(v);
        }
    }
}

// ---------- zero ----------
__global__ __launch_bounds__(256) void zero_kernel(float* __restrict__ p, int n) {
    int i = blockIdx.x * 256 + threadIdx.x;
    if (i < n) p[i] = 0.0f;
}

// ---------- degree / dinv ----------
__global__ __launch_bounds__(256) void deg_count_kernel(const int* __restrict__ dst,
                                                        float* __restrict__ deg, int E) {
    int e = blockIdx.x * 256 + threadIdx.x;
    if (e < E) atomicAdd(&deg[dst[e]], 1.0f);
}
__global__ __launch_bounds__(256) void dinv_kernel(const float* __restrict__ deg,
                                                   float* __restrict__ dinv, int N) {
    int i = blockIdx.x * 256 + threadIdx.x;
    if (i < N) dinv[i] = rsqrtf(deg[i] + 1.0f);
}

// ---------- edge scatter: agg[dst] += ht[src] * dinv[src]*dinv[dst] ----------
__global__ __launch_bounds__(256) void scatter_kernel(const float* __restrict__ ht,
                                                      const int* __restrict__ src,
                                                      const int* __restrict__ dst,
                                                      const float* __restrict__ dinv,
                                                      float* __restrict__ agg, int E) {
    int tid = blockIdx.x * 256 + threadIdx.x;            // 64 threads per edge, 4 feats each
    int e = tid >> 6;
    if (e >= E) return;
    int j = (tid & 63) * 4;
    int s = src[e], d = dst[e];
    float w = dinv[s] * dinv[d];
    const float4 v = *reinterpret_cast<const float4*>(ht + (size_t)s * 256 + j);
    float* base = agg + (size_t)d * 256 + j;
    atomicAdd(base + 0, v.x * w);
    atomicAdd(base + 1, v.y * w);
    atomicAdd(base + 2, v.z * w);
    atomicAdd(base + 3, v.w * w);
}

// ---------- GCN combine: h = relu(agg + ht*selfnorm + bg) ----------
__global__ __launch_bounds__(256) void combine_kernel(const float* __restrict__ agg,
                                                      const float* __restrict__ ht,
                                                      const float* __restrict__ dinv,
                                                      const float* __restrict__ bg,
                                                      unsigned short* __restrict__ h_bf,
                                                      float* __restrict__ h_f32 /*nullable*/) {
    int n = blockIdx.x, j = threadIdx.x;
    float sn = dinv[n]; sn *= sn;
    size_t idx = (size_t)n * 256 + j;
    float v = fmaxf(agg[idx] + ht[idx] * sn + bg[j], 0.0f);
    h_bf[idx] = f2bf(v);
    if (h_f32) h_f32[idx] = v;
}

// ---------- pooling partial reduce (32 rows per block) ----------
__global__ __launch_bounds__(256) void pool_reduce_kernel(const float* __restrict__ h,
                                                          float* __restrict__ sum,
                                                          unsigned int* __restrict__ maxb,
                                                          int N) {
    int j = threadIdx.x;
    float s = 0.0f, m = 0.0f;                            // h >= 0 post-ReLU
    const float* p = h + (size_t)blockIdx.x * 32 * 256 + j;
    #pragma unroll 4
    for (int r = 0; r < 32; ++r) {
        int n = blockIdx.x * 32 + r;
        if (n < N) { float v = p[(size_t)r * 256]; s += v; m = fmaxf(m, v); }
    }
    atomicAdd(&sum[j], s);
    atomicMax(&maxb[j], __float_as_uint(m));             // valid bitwise compare for >= 0 floats
}

// ---------- pool head MLP (single block) ----------
__global__ __launch_bounds__(256) void pool_mlp_kernel(const float* __restrict__ sum,
                                                       const unsigned int* __restrict__ maxb,
                                                       const float* __restrict__ Wp1,
                                                       const float* __restrict__ bp1,
                                                       const float* __restrict__ Wp2,
                                                       const float* __restrict__ bp2,
                                                       float* __restrict__ emb, int N) {
    __shared__ float g[512];
    __shared__ float hid[256];
    int j = threadIdx.x;
    g[j]       = sum[j] / (float)N;
    g[256 + j] = __uint_as_float(maxb[j]);
    __syncthreads();
    float a = bp1[j];
    for (int k = 0; k < 512; ++k) a += g[k] * Wp1[k * 256 + j];
    hid[j] = fmaxf(a, 0.0f);
    __syncthreads();
    if (j < 128) {
        float a2 = bp2[j];
        for (int k = 0; k < 256; ++k) a2 += hid[k] * Wp2[k * 128 + j];
        emb[j] = a2;
    }
}

// ---------- decoder layer 2: [N,256] bf16 @ [256,2] ----------
__global__ __launch_bounds__(256) void dec2_kernel(const unsigned short* __restrict__ a,
                                                   const float* __restrict__ Wd2,
                                                   const float* __restrict__ bd2,
                                                   float* __restrict__ out, int N) {
    __shared__ float w0[256], w1[256];
    int t = threadIdx.x;
    w0[t] = Wd2[t * 2 + 0];
    w1[t] = Wd2[t * 2 + 1];
    __syncthreads();
    int lane = t & 31, wave = t >> 5;
    int n = blockIdx.x * 8 + wave;
    if (n >= N) return;
    const unsigned short* row = a + (size_t)n * 256 + lane * 8;
    float s0 = 0.0f, s1 = 0.0f;
    #pragma unroll
    for (int i = 0; i < 8; ++i) {
        float v = bf2f(row[i]);
        s0 += v * w0[lane * 8 + i];
        s1 += v * w1[lane * 8 + i];
    }
    #pragma unroll
    for (int off = 16; off > 0; off >>= 1) {
        s0 += __shfl_xor(s0, off, 32);
        s1 += __shfl_xor(s1, off, 32);
    }
    if (lane == 0) {
        out[(size_t)n * 2 + 0] = s0 + bd2[0];
        out[(size_t)n * 2 + 1] = s1 + bd2[1];
    }
}

// ---------- host ----------
extern "C" void kernel_launch(void* const* d_in, const int* in_sizes, int n_in,
                              void* d_out, int out_size, void* d_ws, size_t ws_size,
                              hipStream_t stream) {
    const float* x   = (const float*)d_in[0];
    const float* We1 = (const float*)d_in[1];
    const float* be1 = (const float*)d_in[2];
    const float* We2 = (const float*)d_in[3];
    const float* be2 = (const float*)d_in[4];
    const float* Wg  = (const float*)d_in[5];
    const float* bg  = (const float*)d_in[6];
    const float* Wp1 = (const float*)d_in[7];
    const float* bp1 = (const float*)d_in[8];
    const float* Wp2 = (const float*)d_in[9];
    const float* bp2 = (const float*)d_in[10];
    const float* Wd1 = (const float*)d_in[11];
    const float* bd1 = (const float*)d_in[12];
    const float* Wd2 = (const float*)d_in[13];
    const float* bd2 = (const float*)d_in[14];
    const int*   ei  = (const int*)d_in[15];

    const int N = in_sizes[0] / 2;        // x is [N, 2]
    const int E = in_sizes[15] / 2;       // edge_index is [2, E]
    const int* srcIdx = ei;
    const int* dstIdx = ei + E;

    // workspace carve-up (256B aligned)
    char* ws = (char*)d_ws;
    size_t off = 0;
    auto alloc = [&](size_t bytes) -> void* {
        void* p = ws + off;
        off = (off + bytes + 255) & ~(size_t)255;
        return p;
    };
    float*          ws_ht   = (float*)alloc((size_t)N * 256 * 4);
    float*          ws_agg  = (float*)alloc((size_t)N * 256 * 4);
    unsigned short* ws_hbf  = (unsigned short*)alloc((size_t)N * 256 * 2);
    unsigned short* ws_pbf  = (unsigned short*)alloc((size_t)N * 256 * 2);  // enc-pre / dec-hidden
    unsigned short* ws_wbf  = (unsigned short*)alloc((size_t)5 * 65536 * 2);
    float*          ws_deg  = (float*)alloc((size_t)N * 4);
    float*          ws_dinv = (float*)alloc((size_t)N * 4);
    float*          ws_sum  = (float*)alloc(256 * 4);
    unsigned int*   ws_max  = (unsigned int*)alloc(256 * 4);   // contiguous with ws_sum

    float* out_emb = (float*)d_out;                   // [128]
    float* out_h   = out_emb + 128;                   // [N,256]
    float* out_rec = out_h + (size_t)N * 256;         // [N,2]

    const int gemm_blocks = (N + 31) / 32;

    // 1) weights -> swizzled bf16 (L2-resident)
    conv_w_kernel<<<256, 256, 0, stream>>>(We2,              ws_wbf + 0 * 65536);
    conv_w_kernel<<<256, 256, 0, stream>>>(Wg + 0 * 65536,   ws_wbf + 1 * 65536);
    conv_w_kernel<<<256, 256, 0, stream>>>(Wg + 1 * 65536,   ws_wbf + 2 * 65536);
    conv_w_kernel<<<256, 256, 0, stream>>>(Wg + 2 * 65536,   ws_wbf + 3 * 65536);
    conv_w_kernel<<<256, 256, 0, stream>>>(Wd1,              ws_wbf + 4 * 65536);

    // 2) encoder: pre = relu(x@We1+be1) ; h0 = pre @ We2 + be2 (bf16 out)
    enc_pre_kernel<<<N, 256, 0, stream>>>(x, We1, be1, ws_pbf);
    gemm256_wmma_kernel<<<gemm_blocks, 256, 0, stream>>>(ws_pbf, ws_wbf + 0 * 65536, be2,
                                                         nullptr, ws_hbf, N, 0);

    // 3) degrees / normalization
    zero_kernel<<<(N + 255) / 256, 256, 0, stream>>>(ws_deg, N);
    deg_count_kernel<<<(E + 255) / 256, 256, 0, stream>>>(dstIdx, ws_deg, E);
    dinv_kernel<<<(N + 255) / 256, 256, 0, stream>>>(ws_deg, ws_dinv, N);

    // 4) GCN layers
    for (int i = 0; i < 3; ++i) {
        gemm256_wmma_kernel<<<gemm_blocks, 256, 0, stream>>>(ws_hbf, ws_wbf + (1 + i) * 65536,
                                                             nullptr, ws_ht, nullptr, N, 0);
        zero_kernel<<<(N * 256 + 255) / 256, 256, 0, stream>>>(ws_agg, N * 256);
        scatter_kernel<<<(E * 64 + 255) / 256, 256, 0, stream>>>(ws_ht, srcIdx, dstIdx,
                                                                 ws_dinv, ws_agg, E);
        combine_kernel<<<N, 256, 0, stream>>>(ws_agg, ws_ht, ws_dinv, bg + i * 256, ws_hbf,
                                              (i == 2) ? out_h : nullptr);
    }

    // 5) pooling + head MLP
    zero_kernel<<<2, 256, 0, stream>>>(ws_sum, 512);  // zeros sum[256] and maxbits[256]
    pool_reduce_kernel<<<(N + 31) / 32, 256, 0, stream>>>(out_h, ws_sum, ws_max, N);
    pool_mlp_kernel<<<1, 256, 0, stream>>>(ws_sum, ws_max, Wp1, bp1, Wp2, bp2, out_emb, N);

    // 6) decoder: d1 = relu(h @ Wd1 + bd1) (bf16) ; rec = d1 @ Wd2 + bd2
    gemm256_wmma_kernel<<<gemm_blocks, 256, 0, stream>>>(ws_hbf, ws_wbf + 4 * 65536, bd1,
                                                         nullptr, ws_pbf, N, 1);
    dec2_kernel<<<(N + 7) / 8, 256, 0, stream>>>(ws_pbf, Wd2, bd2, out_rec, N);
}
